// MiniTransformerBlock_15350213116410
// MI455X (gfx1250) — matrix operand order repaired and checked by reading, not compiled, not run on hardware
//
#include <hip/hip_runtime.h>

typedef __attribute__((ext_vector_type(16))) __bf16 v16bf;
typedef __attribute__((ext_vector_type(8)))  float  v8f;
typedef unsigned int u32x4 __attribute__((ext_vector_type(4)));
typedef int i32x4 __attribute__((ext_vector_type(4)));
typedef int i32x8 __attribute__((ext_vector_type(8)));

#if __has_builtin(__builtin_amdgcn_tensor_load_to_lds)
#define HAS_TDM 1
#else
#define HAS_TDM 0
#endif

// ---------------- helpers ----------------
__device__ inline unsigned short f2bf(float f) {
  union { float f; unsigned u; } v; v.f = f;
  unsigned r = v.u + 0x7FFFu + ((v.u >> 16) & 1u);   // round-to-nearest-even
  return (unsigned short)(r >> 16);
}
__device__ inline float bf2f(unsigned short s) {
  union { unsigned u; float f; } v; v.u = ((unsigned)s) << 16; return v.f;
}
__device__ inline v8f zero8() {
  v8f z = {0.f,0.f,0.f,0.f,0.f,0.f,0.f,0.f}; return z;
}
__device__ inline v8f wmma_bf16(v16bf a, v16bf b, v8f c) {
  return __builtin_amdgcn_wmma_f32_16x16x32_bf16(false, a, false, b, (short)0, c, false, false);
}

union V16 { v16bf v; u32x4 q[2]; };

#if HAS_TDM
__device__ inline void tdm_wait0() {
#if __has_builtin(__builtin_amdgcn_s_wait_tensorcnt)
  __builtin_amdgcn_s_wait_tensorcnt(0);
#else
  asm volatile("s_wait_tensorcnt 0x0" ::: "memory");
#endif
}

// TDM 2D tile load: 128 rows x 64 bf16 elems, global row stride = K elems,
// LDS dest padded 16B per 128B row (pad_interval=32 DW, pad_amount=4 DW) -> 144B stride.
__device__ inline void tdm_load_tile(const unsigned short* gsrc, unsigned ldsoff, int K) {
  unsigned long long ga = (unsigned long long)(size_t)gsrc;
  u32x4 g0;
  g0[0] = 1u;                                                   // count=1 (valid D#)
  g0[1] = ldsoff;                                               // lds_addr (bytes)
  g0[2] = (unsigned)(ga & 0xFFFFFFFFull);                       // global_addr[31:0]
  g0[3] = (unsigned)((ga >> 32) & 0x1FFFFFFull) | (2u << 30);   // global_addr[56:32] | type=2
  i32x8 g1;
  g1[0] = (int)((1u << 16)     // data_size = 2 bytes
              | (1u << 20)     // pad_enable
              | (4u << 22)     // pad_interval: 32 DWORDs (=128B, one tile row)
              | (3u << 25));   // pad_amount: 4 DWORDs (=16B)
  unsigned d0 = (unsigned)K;   // tensor_dim0
  unsigned d1 = 128u;          // tensor_dim1
  g1[1] = (int)((d0 & 0xFFFFu) << 16);                       // tensor_dim0[15:0]
  g1[2] = (int)((d0 >> 16) | ((d1 & 0xFFFFu) << 16));        // dim0[31:16] | dim1[15:0]
  g1[3] = (int)((d1 >> 16) | (64u << 16));                   // dim1[31:16] | tile_dim0=64
  g1[4] = 128;                                               // tile_dim1=128, tile_dim2=0
  g1[5] = (int)d0;                                           // tensor_dim0_stride lo (=K)
  g1[6] = 0;                                                 // stride hi / dim1_stride lo
  g1[7] = 0;
  i32x4 z4 = {0, 0, 0, 0};
#if defined(__clang_major__) && (__clang_major__ >= 23)
  i32x8 z8 = {0, 0, 0, 0, 0, 0, 0, 0};
  __builtin_amdgcn_tensor_load_to_lds(g0, g1, z4, z4, z8, 0);
#else
  __builtin_amdgcn_tensor_load_to_lds(g0, g1, z4, z4, 0);
#endif
}
#endif  // HAS_TDM

// ---------------- fp32 -> bf16 convert (packed stores) ----------------
__global__ __launch_bounds__(256)
void f2bf_kernel(const float* __restrict__ in, unsigned short* __restrict__ out, size_t n) {
  size_t i = ((size_t)blockIdx.x * 256 + threadIdx.x) * 8;
  if (i + 7 < n) {
    u32x4 o;
    for (int e = 0; e < 4; e++) {
      float lo = in[i + 2 * e], hi = in[i + 2 * e + 1];
      o[e] = (unsigned)f2bf(lo) | ((unsigned)f2bf(hi) << 16);
    }
    *(u32x4*)(out + i) = o;
  }
}

// ---------------- RMSNorm (f32 in -> bf16 out) ----------------
__global__ __launch_bounds__(256)
void rmsnorm_kernel(const float* __restrict__ x, const float* __restrict__ w,
                    unsigned short* __restrict__ out, int H) {
  int row = blockIdx.x;
  const float* xr = x + (size_t)row * H;
  float ss = 0.f;
  for (int i = threadIdx.x; i < H; i += 256) { float v = xr[i]; ss += v * v; }
  __shared__ float red[8];
  for (int off = 16; off; off >>= 1) ss += __shfl_xor(ss, off, 32);
  if ((threadIdx.x & 31) == 0) red[threadIdx.x >> 5] = ss;
  __syncthreads();
  if (threadIdx.x == 0) {
    float v = 0.f;
    for (int i = 0; i < 8; i++) v += red[i];
    red[0] = v;
  }
  __syncthreads();
  float rinv = rsqrtf(red[0] / (float)H + 1e-6f);
  unsigned short* orow = out + (size_t)row * H;
  for (int i = threadIdx.x; i < H; i += 256) orow[i] = f2bf(xr[i] * rinv * w[i]);
}

// ---------------- GEMM: C[M,N] = A[M,K](bf16) * W[N,K]^T(bf16) ----------------
// TDM double-buffered tiles, K-step 64 (16 WMMA per stage per wave).
// OUTF32 == 0: bf16 output; OUTF32 == 1: f32 output + residual add
template<int OUTF32>
__global__ __launch_bounds__(256)
void gemm_kernel(const unsigned short* __restrict__ A,
                 const unsigned short* __restrict__ W,
                 void* __restrict__ Cout,
                 const float* __restrict__ residual,
                 int M, int N, int K) {
  __shared__ __align__(16) unsigned short lA[2][128][72];  // 128x64 tile + 8-short row pad
  __shared__ __align__(16) unsigned short lB[2][128][72];

  const int tid  = threadIdx.x;
  const int wave = tid >> 5;
  const int lane = tid & 31;
  const int half = lane >> 4;
  const int l16  = lane & 15;
  const int m0 = blockIdx.y * 128;
  const int n0 = blockIdx.x * 128;
  const int wm = (wave >> 1) * 32;   // 0,32,64,96
  const int wn = (wave & 1) * 64;    // 0,64

  v8f acc[2][4];
  for (int i = 0; i < 2; i++)
    for (int j = 0; j < 4; j++) acc[i][j] = zero8();

  const int KT = K >> 6;

#if HAS_TDM
  if (wave == 0) {
    tdm_load_tile(A + (size_t)m0 * K, (unsigned)(size_t)&lA[0][0][0], K);
    tdm_load_tile(W + (size_t)n0 * K, (unsigned)(size_t)&lB[0][0][0], K);
  }
#endif

  for (int kt = 0; kt < KT; kt++) {
    const int buf = kt & 1;
#if HAS_TDM
    if (wave == 0) tdm_wait0();       // current buf tiles landed
    __syncthreads();                  // visible to all; prior reads of other buf done
    if (wave == 0 && kt + 1 < KT) {   // overlap next-tile DMA with compute
      const int k0n = (kt + 1) * 64;
      tdm_load_tile(A + (size_t)m0 * K + k0n, (unsigned)(size_t)&lA[buf ^ 1][0][0], K);
      tdm_load_tile(W + (size_t)n0 * K + k0n, (unsigned)(size_t)&lB[buf ^ 1][0][0], K);
    }
#else
    {
      const int lrow  = tid >> 1;
      const int lcolb = (tid & 1) * 32;
      const int k0 = kt * 64;
      const u32x4* gA = (const u32x4*)(A + (size_t)(m0 + lrow) * K + k0 + lcolb);
      const u32x4* gW = (const u32x4*)(W + (size_t)(n0 + lrow) * K + k0 + lcolb);
      for (int e = 0; e < 4; e++) {
        *(u32x4*)&lA[buf][lrow][lcolb + e * 8] = gA[e];
        *(u32x4*)&lB[buf][lrow][lcolb + e * 8] = gW[e];
      }
    }
    __syncthreads();
#endif

    for (int kc = 0; kc < 2; kc++) {
      V16 af[2], bfm[4];
      for (int i = 0; i < 2; i++) {
        int arow = wm + i * 16 + l16;
        af[i].q[0] = *(const u32x4*)&lA[buf][arow][kc * 32 + half * 8];
        af[i].q[1] = *(const u32x4*)&lA[buf][arow][kc * 32 + 16 + half * 8];
      }
      for (int j = 0; j < 4; j++) {
        int brow = wn + j * 16 + l16;
        bfm[j].q[0] = *(const u32x4*)&lB[buf][brow][kc * 32 + half * 8];
        bfm[j].q[1] = *(const u32x4*)&lB[buf][brow][kc * 32 + 16 + half * 8];
      }
      for (int i = 0; i < 2; i++)
        for (int j = 0; j < 4; j++)
          acc[i][j] = wmma_bf16(af[i].v, bfm[j].v, acc[i][j]);
    }
#if !HAS_TDM
    __syncthreads();
#endif
  }

  // C/D layout: VGPR r -> row (r | r+8 by lane half), col = lane%16
  for (int i = 0; i < 2; i++) {
    for (int j = 0; j < 4; j++) {
      int rowb = m0 + wm + i * 16 + half * 8;
      int col  = n0 + wn + j * 16 + l16;
      for (int r = 0; r < 8; r++) {
        size_t idx = (size_t)(rowb + r) * N + col;
        float val = acc[i][j][r];
        if (OUTF32) {
          float* C = (float*)Cout;
          C[idx] = residual[idx] + val;
        } else {
          unsigned short* C = (unsigned short*)Cout;
          C[idx] = f2bf(val);
        }
      }
    }
  }
}

// ---------------- Flash attention (bf16 q,k,v -> bf16 out) ----------------
// Grid: (T/128, NUM_HEADS, B); block: 256 (8 waves, each owns 16 q rows)
__global__ __launch_bounds__(256)
void flash_attn_kernel(const unsigned short* __restrict__ Q,
                       const unsigned short* __restrict__ Kb,
                       const unsigned short* __restrict__ Vb,
                       unsigned short* __restrict__ O) {
  const int T = 2048, H = 2048, HD = 128;
  const int qt = blockIdx.x;
  const int h  = blockIdx.y;
  const int b  = blockIdx.z;
  const int tid  = threadIdx.x;
  const int wave = tid >> 5;
  const int lane = tid & 31;
  const int half = lane >> 4;
  const int l16  = lane & 15;
  const size_t baserow = (size_t)b * T;
  const int coff = h * HD;

  __shared__ __align__(16) unsigned short lK [32][HD + 8];   // keys x dims
  __shared__ __align__(16) unsigned short lVT[HD][32 + 8];   // dims x keys (V^T)
  __shared__ __align__(16) unsigned short lP [8][16][40];    // per-wave P 16x32

  const int qr0 = qt * 128 + wave * 16;

  V16 qf[4];
  for (int kc = 0; kc < 4; kc++) {
    const unsigned short* qrow = Q + (baserow + qr0 + l16) * H + coff + kc * 32;
    qf[kc].q[0] = *(const u32x4*)(qrow + half * 8);
    qf[kc].q[1] = *(const u32x4*)(qrow + 16 + half * 8);
  }

  v8f o[8];
  for (int j = 0; j < 8; j++) o[j] = zero8();
  float mstat[8], lstat[8];
  for (int r = 0; r < 8; r++) { mstat[r] = -1e30f; lstat[r] = 0.f; }
  const float scale = 0.0883883476483184f;   // 1/sqrt(128)

  for (int j0 = 0; j0 < T; j0 += 32) {
    {
      int r = tid >> 3;            // 0..31 (key)
      int c = (tid & 7) * 16;      // 0..112 (dim)
      const u32x4* ksrc = (const u32x4*)(Kb + (baserow + j0 + r) * H + coff + c);
      u32x4 k0 = ksrc[0], k1 = ksrc[1];
      *(u32x4*)&lK[r][c]     = k0;
      *(u32x4*)&lK[r][c + 8] = k1;
      const u32x4* vsrc = (const u32x4*)(Vb + (baserow + j0 + r) * H + coff + c);
      u32x4 v0 = vsrc[0], v1 = vsrc[1];
      unsigned short tmp[16];
      *(u32x4*)&tmp[0] = v0; *(u32x4*)&tmp[8] = v1;
      for (int e = 0; e < 16; e++) lVT[c + e][r] = tmp[e];
    }
    __syncthreads();

    v8f s0 = zero8(), s1 = zero8();
    for (int kc = 0; kc < 4; kc++) {
      V16 kf0, kf1;
      kf0.q[0] = *(const u32x4*)&lK[l16][kc * 32 + half * 8];
      kf0.q[1] = *(const u32x4*)&lK[l16][kc * 32 + 16 + half * 8];
      kf1.q[0] = *(const u32x4*)&lK[16 + l16][kc * 32 + half * 8];
      kf1.q[1] = *(const u32x4*)&lK[16 + l16][kc * 32 + 16 + half * 8];
      s0 = wmma_bf16(qf[kc].v, kf0.v, s0);
      s1 = wmma_bf16(qf[kc].v, kf1.v, s1);
    }

    float corr[8];
    for (int r = 0; r < 8; r++) {
      float mx = fmaxf(s0[r], s1[r]);
      for (int off = 8; off; off >>= 1) mx = fmaxf(mx, __shfl_xor(mx, off, 32));
      mx *= scale;
      float mnew = fmaxf(mstat[r], mx);
      corr[r] = __expf(mstat[r] - mnew);
      mstat[r] = mnew;
      float p0 = __expf(s0[r] * scale - mnew);
      float p1 = __expf(s1[r] * scale - mnew);
      s0[r] = p0; s1[r] = p1;
      float rs = p0 + p1;
      for (int off = 8; off; off >>= 1) rs += __shfl_xor(rs, off, 32);
      lstat[r] = lstat[r] * corr[r] + rs;
    }

    for (int r = 0; r < 8; r++) {
      lP[wave][r + half * 8][l16]      = f2bf(s0[r]);
      lP[wave][r + half * 8][16 + l16] = f2bf(s1[r]);
    }
    asm volatile("s_wait_dscnt 0x0" ::: "memory");
    V16 pf;
    pf.q[0] = *(const u32x4*)&lP[wave][l16][half * 8];
    pf.q[1] = *(const u32x4*)&lP[wave][l16][16 + half * 8];

    for (int nt = 0; nt < 8; nt++)
      for (int r = 0; r < 8; r++) o[nt][r] *= corr[r];
    for (int nt = 0; nt < 8; nt++) {
      V16 vf;
      vf.q[0] = *(const u32x4*)&lVT[nt * 16 + l16][half * 8];
      vf.q[1] = *(const u32x4*)&lVT[nt * 16 + l16][16 + half * 8];
      o[nt] = wmma_bf16(pf.v, vf.v, o[nt]);
    }
    __syncthreads();
  }

  for (int nt = 0; nt < 8; nt++) {
    for (int r = 0; r < 8; r++) {
      int row = qr0 + r + half * 8;
      int col = coff + nt * 16 + l16;
      O[(baserow + row) * H + col] = f2bf(o[nt][r] / lstat[r]);
    }
  }
}

// ---------------- SiLU(gate) * up -> act (bf16) ----------------
__global__ __launch_bounds__(256)
void silumul_kernel(const unsigned short* __restrict__ gate,
                    const unsigned short* __restrict__ up,
                    unsigned short* __restrict__ act, size_t n) {
  size_t i = ((size_t)blockIdx.x * 256 + threadIdx.x) * 4;
  if (i + 3 < n) {
    for (int e = 0; e < 4; e++) {
      float g = bf2f(gate[i + e]);
      float u = bf2f(up[i + e]);
      float s = g / (1.f + __expf(-g));
      act[i + e] = f2bf(s * u);
    }
  }
}

// ---------------- launch ----------------
extern "C" void kernel_launch(void* const* d_in, const int* in_sizes, int n_in,
                              void* d_out, int out_size, void* d_ws, size_t ws_size,
                              hipStream_t stream) {
  const float* x    = (const float*)d_in[0];
  const float* wln1 = (const float*)d_in[1];
  const float* wq   = (const float*)d_in[2];
  const float* wk   = (const float*)d_in[3];
  const float* wv   = (const float*)d_in[4];
  const float* wo   = (const float*)d_in[5];
  const float* wln2 = (const float*)d_in[6];
  const float* wg   = (const float*)d_in[7];
  const float* wu   = (const float*)d_in[8];
  const float* wd   = (const float*)d_in[9];
  float* out = (float*)d_out;

  const int B = 2, T = 2048, H = 2048, I = 8192;
  const int M = B * T;   // 4096

  char* ws = (char*)d_ws;
  size_t off = 0;
  auto alloc = [&](size_t bytes) -> void* {
    void* p = ws + off;
    off += (bytes + 255) & ~(size_t)255;
    return p;
  };
  unsigned short* Wb   = (unsigned short*)alloc((size_t)I * H * 2);  // shared weight slot
  unsigned short* h1   = (unsigned short*)alloc((size_t)M * H * 2);
  unsigned short* qb   = (unsigned short*)alloc((size_t)M * H * 2);
  unsigned short* kb   = (unsigned short*)alloc((size_t)M * H * 2);
  unsigned short* vb   = (unsigned short*)alloc((size_t)M * H * 2);
  unsigned short* attn = (unsigned short*)alloc((size_t)M * H * 2);
  float*          x2   = (float*)alloc((size_t)M * H * 4);
  unsigned short* h2   = (unsigned short*)alloc((size_t)M * H * 2);
  unsigned short* gate = (unsigned short*)alloc((size_t)M * I * 2);
  unsigned short* up   = (unsigned short*)alloc((size_t)M * I * 2);

  dim3 blk(256);
  const size_t nHH = (size_t)H * H, nIH = (size_t)I * H;
  const int gHH = (int)((nHH / 8 + 255) / 256);
  const int gIH = (int)((nIH / 8 + 255) / 256);

  // ln1 + QKV projections
  rmsnorm_kernel<<<M, blk, 0, stream>>>(x, wln1, h1, H);
  f2bf_kernel<<<gHH, blk, 0, stream>>>(wq, Wb, nHH);
  gemm_kernel<0><<<dim3(H / 128, M / 128), blk, 0, stream>>>(h1, Wb, qb, nullptr, M, H, H);
  f2bf_kernel<<<gHH, blk, 0, stream>>>(wk, Wb, nHH);
  gemm_kernel<0><<<dim3(H / 128, M / 128), blk, 0, stream>>>(h1, Wb, kb, nullptr, M, H, H);
  f2bf_kernel<<<gHH, blk, 0, stream>>>(wv, Wb, nHH);
  gemm_kernel<0><<<dim3(H / 128, M / 128), blk, 0, stream>>>(h1, Wb, vb, nullptr, M, H, H);

  // attention + output projection with residual
  flash_attn_kernel<<<dim3(T / 128, 16, B), blk, 0, stream>>>(qb, kb, vb, attn);
  f2bf_kernel<<<gHH, blk, 0, stream>>>(wo, Wb, nHH);
  gemm_kernel<1><<<dim3(H / 128, M / 128), blk, 0, stream>>>(attn, Wb, x2, x, M, H, H);

  // ln2 + SwiGLU MLP with residual
  rmsnorm_kernel<<<M, blk, 0, stream>>>(x2, wln2, h2, H);
  f2bf_kernel<<<gIH, blk, 0, stream>>>(wg, Wb, nIH);
  gemm_kernel<0><<<dim3(I / 128, M / 128), blk, 0, stream>>>(h2, Wb, gate, nullptr, M, I, H);
  f2bf_kernel<<<gIH, blk, 0, stream>>>(wu, Wb, nIH);
  gemm_kernel<0><<<dim3(I / 128, M / 128), blk, 0, stream>>>(h2, Wb, up, nullptr, M, I, H);
  const size_t nMI = (size_t)M * I;
  silumul_kernel<<<(int)((nMI / 4 + 255) / 256), blk, 0, stream>>>(gate, up, gate, nMI);
  f2bf_kernel<<<gIH, blk, 0, stream>>>(wd, Wb, nIH);
  gemm_kernel<1><<<dim3(H / 128, M / 128), blk, 0, stream>>>(gate, Wb, out, x2, M, H, I);
}
